// BertAttention_20504173871698
// MI455X (gfx1250) — matrix operand order repaired and checked by reading, not compiled, not run on hardware
//
#include <hip/hip_runtime.h>
#include <hip/hip_bf16.h>
#include <stdint.h>

static constexpr int kB   = 4;
static constexpr int kSQ  = 2048;
static constexpr int kSK  = 2048;
static constexpr int kHID = 1024;
static constexpr int kNH  = 16;
static constexpr int kHD  = 64;

typedef __attribute__((ext_vector_type(16))) __bf16 v16bf;
typedef __attribute__((ext_vector_type(8)))  float  v8f;
typedef __attribute__((ext_vector_type(4)))  unsigned int v4u;
typedef __attribute__((ext_vector_type(8)))  int v8i;
typedef __attribute__((ext_vector_type(4)))  int v4i;

struct U128 { uint32_t x, y, z, w; };
struct F128 { float x, y, z, w; };

union Frag16 {
    v16bf v;
    U128  q[2];
    unsigned short h[16];
};

__device__ __forceinline__ unsigned short f2bf(float f) {
    uint32_t u = __float_as_uint(f);
    u += 0x7FFFu + ((u >> 16) & 1u);   // round-to-nearest-even
    return (unsigned short)(u >> 16);
}

__device__ __forceinline__ v8f bfwmma(const Frag16& a, const Frag16& b, v8f c) {
    return __builtin_amdgcn_wmma_f32_16x16x32_bf16(
        false, a.v, false, b.v, (short)0, c, false, false);
}

// ---------------------------------------------------------------------------
// TDM: load a 2D bf16 tile (tile0 elems/row contiguous, tile1 rows, row pitch
// stride0 elems) from global into LDS at lds_addr (packed, row-major).
// D# layout per CDNA5 ISA ch.8 (group0 128b, group1 256b; groups 2/3 unused).
// ---------------------------------------------------------------------------
__device__ __forceinline__ void tdm_load_2d(unsigned lds_addr, const void* gaddr,
                                            unsigned tile0, unsigned tile1,
                                            unsigned stride0) {
    const uint64_t ga = (uint64_t)(uintptr_t)gaddr;
    v4u g0;
    g0[0] = 1u;                                   // count=1 (valid), user mode
    g0[1] = lds_addr;                             // LDS byte address
    g0[2] = (unsigned)(ga & 0xFFFFFFFFu);         // global_addr[31:0]
    g0[3] = (unsigned)((ga >> 32) & 0x01FFFFFFu)  // global_addr[56:32]
          | (2u << 30);                           // type=2 ("image")
    v8i g1;
    g1[0] = (int)(1u << 16);                      // data_size=1 -> 2 bytes
    g1[1] = (int)((tile0 & 0xFFFFu) << 16);       // tensor_dim0[15:0] = tile0
    g1[2] = (int)((tile1 & 0xFFFFu) << 16);       // dim0 hi=0 | tensor_dim1 lo
    g1[3] = (int)(tile0 << 16);                   // dim1 hi=0 | tile_dim0
    g1[4] = (int)(tile1 & 0xFFFFu);               // tile_dim1 | tile_dim2=0
    g1[5] = (int)stride0;                         // tensor_dim0_stride lo32
    g1[6] = 0;                                    // stride0 hi | stride1 lo
    g1[7] = 0;                                    // stride1 hi
    v4i z4 = {};
#if __clang_major__ >= 23
    v8i z8 = {};
    __builtin_amdgcn_tensor_load_to_lds(g0, g1, z4, z4, z8, 0);
#else
    __builtin_amdgcn_tensor_load_to_lds(g0, g1, z4, z4, 0);
#endif
}

// LDS byte offset of a shared-memory pointer (generic LDS ptr: low 32 bits)
__device__ __forceinline__ unsigned lds_off(const void* p) {
    return (unsigned)(uintptr_t)p;
}

// ---------------------------------------------------------------------------
// Kernel 1: QKV projections, fp32 -> bf16.
//   z=0: Q = hidden  @ Wq + bq -> Qb [B*SQ, HID] row-major
//   z=1: K = context @ Wk + bk -> Kb [B*SK, HID] row-major
//   z=2: V = context @ Wv + bv -> Vt [B, NH, HD, SK]  (transposed for P@V)
// ---------------------------------------------------------------------------
__global__ __launch_bounds__(256) void qkv_gemm(
    const float* __restrict__ hidden, const float* __restrict__ context,
    const float* __restrict__ Wq, const float* __restrict__ bq,
    const float* __restrict__ Wk, const float* __restrict__ bk,
    const float* __restrict__ Wv, const float* __restrict__ bv,
    unsigned short* __restrict__ Qb, unsigned short* __restrict__ Kb,
    unsigned short* __restrict__ Vt)
{
    const int zz = blockIdx.z;
    const float* X    = (zz == 0) ? hidden : context;
    const float* W    = (zz == 0) ? Wq : (zz == 1) ? Wk : Wv;
    const float* bias = (zz == 0) ? bq : (zz == 1) ? bk : bv;

    __shared__ __align__(16) unsigned short As[128][32];   // 8 KB
    __shared__ __align__(16) unsigned short BsT[64][32];   // 4 KB (transposed)

    const int tid  = threadIdx.x;
    const int lane = tid & 31;
    const int wave = tid >> 5;
    const int m0 = blockIdx.y * 128;
    const int n0 = blockIdx.x * 64;
    const int wm = (wave & 3) * 32;
    const int wn = (wave >> 2) * 32;
    const int kh = lane >> 4;
    const int ln = lane & 15;

    v8f acc[2][2] = {};

    const int arow = tid >> 1;
    const int ac0  = (tid & 1) * 16;
    const int brow = tid >> 3;
    const int bc0  = (tid & 7) * 8;

    for (int k0 = 0; k0 < kHID; k0 += 32) {
        __syncthreads();
        {   // A tile 128x32 (fp32 -> bf16)
            const float* src = X + (size_t)(m0 + arow) * kHID + k0 + ac0;
            #pragma unroll
            for (int j = 0; j < 16; j += 4) {
                F128 f = *(const F128*)(src + j);
                As[arow][ac0 + j + 0] = f2bf(f.x);
                As[arow][ac0 + j + 1] = f2bf(f.y);
                As[arow][ac0 + j + 2] = f2bf(f.z);
                As[arow][ac0 + j + 3] = f2bf(f.w);
            }
        }
        {   // B tile 32x64, transposed in LDS
            const float* src = W + (size_t)(k0 + brow) * kHID + n0 + bc0;
            #pragma unroll
            for (int j = 0; j < 8; j += 4) {
                F128 f = *(const F128*)(src + j);
                BsT[bc0 + j + 0][brow] = f2bf(f.x);
                BsT[bc0 + j + 1][brow] = f2bf(f.y);
                BsT[bc0 + j + 2][brow] = f2bf(f.z);
                BsT[bc0 + j + 3][brow] = f2bf(f.w);
            }
        }
        __syncthreads();

        Frag16 af[2], bfm[2];
        #pragma unroll
        for (int i = 0; i < 2; ++i) {
            const int row = wm + i * 16 + ln;
            af[i].q[0] = *(const U128*)&As[row][kh * 8];
            af[i].q[1] = *(const U128*)&As[row][16 + kh * 8];
        }
        #pragma unroll
        for (int j = 0; j < 2; ++j) {
            const int col = wn + j * 16 + ln;
            bfm[j].q[0] = *(const U128*)&BsT[col][kh * 8];
            bfm[j].q[1] = *(const U128*)&BsT[col][16 + kh * 8];
        }
        #pragma unroll
        for (int i = 0; i < 2; ++i)
            #pragma unroll
            for (int j = 0; j < 2; ++j)
                acc[i][j] = bfwmma(af[i], bfm[j], acc[i][j]);
    }

    #pragma unroll
    for (int i = 0; i < 2; ++i) {
        #pragma unroll
        for (int j = 0; j < 2; ++j) {
            const int coln = n0 + wn + j * 16 + ln;
            const float bb = bias[coln];
            #pragma unroll
            for (int r = 0; r < 8; ++r) {
                const int rowm = m0 + wm + i * 16 + kh * 8 + r;
                const unsigned short val = f2bf(acc[i][j][r] + bb);
                if (zz == 2) {
                    const int bi = rowm >> 11, kk = rowm & 2047;
                    const int hh = coln >> 6,  dd = coln & 63;
                    Vt[((((size_t)bi * kNH + hh) * kHD + dd) << 11) + kk] = val;
                } else {
                    unsigned short* dst = (zz == 0) ? Qb : Kb;
                    dst[(size_t)rowm * kHID + coln] = val;
                }
            }
        }
    }
}

// ---------------------------------------------------------------------------
// Kernel 2: flash attention. Block = 4 waves = 64 queries of one (b,h).
// K/V tiles staged into LDS by the Tensor Data Mover, double-buffered,
// overlapped with WMMA compute. 8 WMMAs per wave per 32-key tile.
// ---------------------------------------------------------------------------
__global__ __launch_bounds__(128) void flash_attn(
    const unsigned short* __restrict__ Qb,
    const unsigned short* __restrict__ Kb,
    const unsigned short* __restrict__ Vt,
    float* __restrict__ out)
{
    __shared__ __align__(16) unsigned short Ks[2][32][64];    // [buf][key][d] 8KB
    __shared__ __align__(16) unsigned short Vs[2][64][32];    // [buf][d][key] 8KB
    __shared__ __align__(16) unsigned short Pls[4][16][32];   // per-wave P  4KB

    const int tid  = threadIdx.x;
    const int lane = tid & 31;
    const int w    = tid >> 5;
    const int bid  = blockIdx.x;
    const int qb   = bid & 31;           // SQ/64 = 32 query blocks
    const int h    = (bid >> 5) & 15;
    const int b    = bid >> 9;
    const int kh = lane >> 4;
    const int ln = lane & 15;

    const size_t kgbase = ((size_t)b * kSK) * kHID + h * kHD;          // elems
    const size_t vgbase = (((size_t)b * kNH + h) * kHD) * kSK;         // elems

    // Q A-fragments for this wave's 16-query tile
    Frag16 qf[2];
    {
        const int qrow = qb * 64 + w * 16 + ln;
        const size_t base = ((size_t)(b * kSQ + qrow)) * kHID + h * kHD;
        #pragma unroll
        for (int ks = 0; ks < 2; ++ks) {
            qf[ks].q[0] = *(const U128*)(Qb + base + ks * 32 + kh * 8);
            qf[ks].q[1] = *(const U128*)(Qb + base + ks * 32 + 16 + kh * 8);
        }
    }

    float mrow[8], srow[8];
    #pragma unroll
    for (int r = 0; r < 8; ++r) { mrow[r] = -3.0e38f; srow[r] = 0.0f; }
    v8f o[4] = {};
    const float sc = 0.125f;   // 1/sqrt(HD)

    constexpr int NT = kSK / 32;   // 64 key tiles

    // Prologue: TDM-load tile 0 into buffer 0 (one op each from wave 0)
    if (w == 0) {
        tdm_load_2d(lds_off(&Ks[0][0][0]), Kb + kgbase, kHD, 32, kHID);
        tdm_load_2d(lds_off(&Vs[0][0][0]), Vt + vgbase, 32, kHD, kSK);
    }

    for (int kt = 0; kt < NT; ++kt) {
        const int buf = kt & 1;

        if (w == 0) __builtin_amdgcn_s_wait_tensorcnt(0);   // tiles[buf] done
        __syncthreads();                                    // visible to all

        // Prefetch next tile into the other buffer (overlap with compute)
        if (w == 0 && kt + 1 < NT) {
            const int kn = (kt + 1) * 32;
            tdm_load_2d(lds_off(&Ks[buf ^ 1][0][0]),
                        Kb + kgbase + (size_t)kn * kHID, kHD, 32, kHID);
            tdm_load_2d(lds_off(&Vs[buf ^ 1][0][0]),
                        Vt + vgbase + kn, 32, kHD, kSK);
        }

        // Scores: Q(16x64) x K^T(64x32) as 2 key-subtiles x 2 k-steps
        v8f s0 = {}, s1 = {};
        #pragma unroll
        for (int ks = 0; ks < 2; ++ks) {
            Frag16 k0f, k1f;
            const unsigned short* r0 = &Ks[buf][ln][ks * 32 + kh * 8];
            const unsigned short* r1 = &Ks[buf][16 + ln][ks * 32 + kh * 8];
            k0f.q[0] = *(const U128*)r0;  k0f.q[1] = *(const U128*)(r0 + 16);
            k1f.q[0] = *(const U128*)r1;  k1f.q[1] = *(const U128*)(r1 + 16);
            s0 = bfwmma(qf[ks], k0f, s0);
            s1 = bfwmma(qf[ks], k1f, s1);
        }

        // Online softmax over 32 keys; rows live in 16-lane halves
        #pragma unroll
        for (int r = 0; r < 8; ++r) {
            const float v0 = s0[r] * sc, v1 = s1[r] * sc;
            float rm = fmaxf(v0, v1);
            #pragma unroll
            for (int msk = 1; msk < 16; msk <<= 1)
                rm = fmaxf(rm, __shfl_xor(rm, msk, 32));
            const float mnew = fmaxf(mrow[r], rm);
            const float p0 = __expf(v0 - mnew);
            const float p1 = __expf(v1 - mnew);
            float rs = p0 + p1;
            #pragma unroll
            for (int msk = 1; msk < 16; msk <<= 1)
                rs += __shfl_xor(rs, msk, 32);
            const float corr = __expf(mrow[r] - mnew);
            srow[r] = srow[r] * corr + rs;
            mrow[r] = mnew;
            #pragma unroll
            for (int t = 0; t < 4; ++t) o[t][r] *= corr;
            const int prow = kh * 8 + r;
            Pls[w][prow][ln]      = f2bf(p0);
            Pls[w][prow][16 + ln] = f2bf(p1);
        }

        asm volatile("s_wait_dscnt 0x0" ::: "memory");   // per-wave P ready

        Frag16 pf;
        pf.q[0] = *(const U128*)&Pls[w][ln][kh * 8];
        pf.q[1] = *(const U128*)&Pls[w][ln][16 + kh * 8];

        // O += P(16x32) x V(32x64), V from LDS (contiguous along key dim)
        #pragma unroll
        for (int t = 0; t < 4; ++t) {
            Frag16 vf;
            const unsigned short* vr = &Vs[buf][t * 16 + ln][kh * 8];
            vf.q[0] = *(const U128*)vr;
            vf.q[1] = *(const U128*)(vr + 16);
            o[t] = bfwmma(pf, vf, o[t]);
        }
    }

    // Normalize and store fp32 output [B, SQ, NH*HD]
    #pragma unroll
    for (int t = 0; t < 4; ++t) {
        #pragma unroll
        for (int r = 0; r < 8; ++r) {
            const int qrow = qb * 64 + w * 16 + kh * 8 + r;
            const int col  = h * kHD + t * 16 + ln;
            out[((size_t)(b * kSQ + qrow)) * kHID + col] = o[t][r] / srow[r];
        }
    }
}

extern "C" void kernel_launch(void* const* d_in, const int* in_sizes, int n_in,
                              void* d_out, int out_size, void* d_ws, size_t ws_size,
                              hipStream_t stream) {
    const float* hidden  = (const float*)d_in[0];
    const float* context = (const float*)d_in[1];
    const float* Wq = (const float*)d_in[2];
    const float* bq = (const float*)d_in[3];
    const float* Wk = (const float*)d_in[4];
    const float* bk = (const float*)d_in[5];
    const float* Wv = (const float*)d_in[6];
    const float* bv = (const float*)d_in[7];

    const size_t elems = (size_t)kB * kSQ * kHID;     // 8M bf16 each
    unsigned short* Qb = (unsigned short*)d_ws;
    unsigned short* Kb = Qb + elems;
    unsigned short* Vt = Kb + elems;                  // 48 MB total

    dim3 g1(kHID / 64, (kB * kSQ) / 128, 3);
    qkv_gemm<<<g1, dim3(256), 0, stream>>>(hidden, context, Wq, bq, Wk, bk,
                                           Wv, bv, Qb, Kb, Vt);

    dim3 g2(kB * kNH * (kSQ / 64));   // 2048 blocks, 4 waves each
    flash_attn<<<g2, dim3(128), 0, stream>>>(Qb, Kb, Vt, (float*)d_out);
}
